// Transformer_Model_38878043963939
// MI455X (gfx1250) — matrix operand order repaired and checked by reading, not compiled, not run on hardware
//
#include <hip/hip_runtime.h>
#include <hip/hip_bf16.h>
#include <math.h>

// ---- model constants (match reference) ----
static constexpr int V_  = 32000;
static constexpr int D_  = 1024;
static constexpr int H_  = 16;
static constexpr int HD_ = 64;
static constexpr int L_  = 4;
static constexpr int S_  = 1024;
static constexpr int B_  = 2;
static constexpr int NF_ = 50;
static constexpr int DL_ = 512;

typedef __attribute__((ext_vector_type(16))) _Float16 v16h;
typedef __attribute__((ext_vector_type(8)))  float    v8f;

__device__ __forceinline__ v8f wmma32(v16h a, v16h b, v8f c) {
  // D = A(16x32 f16) * B(32x16 f16) + C(16x16 f32)
  return __builtin_amdgcn_wmma_f32_16x16x32_f16(false, a, false, b, (short)0, c,
                                                false, false);
}

// A-matrix fragment (16x32 f16), rows = row0..row0+15, K = k0..k0+31.
// Per-lane swizzle (wave32): row = lane&15; K(e) = (lane[4]<<4) | e
template <bool ADD2>
__device__ __forceinline__ v16h load_a_frag(const float* __restrict__ A,
                                            const float* __restrict__ A2,
                                            int lda, int row0, int k0) {
  const int Ln = threadIdx.x & 31;
  const int m = Ln & 15, hi = (Ln >> 4) & 1;
  const float* p = A + (size_t)(row0 + m) * lda + k0;
  const float* p2 = ADD2 ? (A2 + (size_t)(row0 + m) * lda + k0) : nullptr;
  v16h a;
#pragma unroll
  for (int e = 0; e < 16; ++e) {
    const int K = (hi << 4) | e;
    float v = p[K];
    if (ADD2) v += p2[K];
    a[e] = (_Float16)v;
  }
  return a;
}

// B-matrix fragment (32x16 f16) from row-major W[K x N]:
// col = col0 + (lane&15); K(e) = (lane[4]<<4) | e
__device__ __forceinline__ v16h load_b_frag(const float* __restrict__ W,
                                            int ldb, int k0, int col0) {
  const int Ln = threadIdx.x & 31;
  const int n = Ln & 15, hi = (Ln >> 4) & 1;
  v16h b;
#pragma unroll
  for (int e = 0; e < 16; ++e) {
    const int K = (hi << 4) | e;
    b[e] = (_Float16)W[(size_t)(k0 + K) * ldb + (col0 + n)];
  }
  return b;
}

// Transposed B fragment: logical B[kk][nn] = Src[(n0+nn)*ld + k0+kk]  (for Q·K^T)
__device__ __forceinline__ v16h load_bT_frag(const float* __restrict__ Src,
                                             int ld, int n0, int k0) {
  const int Ln = threadIdx.x & 31;
  const int n = Ln & 15, hi = (Ln >> 4) & 1;
  v16h b;
#pragma unroll
  for (int e = 0; e < 16; ++e) {
    const int kk = (hi << 4) | e;
    b[e] = (_Float16)Src[(size_t)(n0 + n) * ld + k0 + kk];
  }
  return b;
}

// ---------------- Fourier embedding base = fourier @ proj_w + proj_b ----------
__global__ __launch_bounds__(256) void embed_base_kernel(
    const int* __restrict__ src, const float* __restrict__ a_n,
    const float* __restrict__ b_n, const float* __restrict__ proj_w,
    const float* __restrict__ proj_b, float* __restrict__ base) {
  const int row = blockIdx.x;          // 0..B*S-1
  const int tok = src[row];
  __shared__ float f[NF_];
  const int t = threadIdx.x;
  if (t < NF_) {
    const float x = (float)tok / (float)V_;
    const float freq = (float)(t + 1);                    // linspace(1, NF, NF)
    const float ang = 6.283185307179586f * freq * x;
    f[t] = a_n[(size_t)tok * NF_ + t] * cosf(ang) +
           b_n[(size_t)tok * NF_ + t] * sinf(ang);
  }
  __syncthreads();
  for (int d = t; d < D_; d += 256) {
    float s = proj_b[d];
#pragma unroll 5
    for (int nn = 0; nn < NF_; ++nn) s += f[nn] * proj_w[(size_t)nn * D_ + d];
    base[(size_t)row * D_ + d] = s;
  }
}

// ---------------- Generic WMMA GEMM: C = A @ W (+epilogue) --------------------
// Block: 256 thr = 8 waves (2x4); wave tile 32x32 -> block tile 64(M) x 128(N).
// EPI: 0 none, 1 add sinusoidal PE (pos = row % S), 2 add bias[col].
template <int EPI, bool ADD2>
__global__ __launch_bounds__(256) void wmma_gemm(
    const float* __restrict__ A, const float* __restrict__ A2,
    const float* __restrict__ W, const float* __restrict__ bias,
    float* __restrict__ C, int N, int K) {
  const int Ln = threadIdx.x & 31;
  const int wv = threadIdx.x >> 5;
  const int wm = wv >> 2, wn = wv & 3;
  const int row0 = blockIdx.y * 64 + wm * 32;
  const int col0 = blockIdx.x * 128 + wn * 32;
  const int n = Ln & 15, hi = (Ln >> 4) & 1;

  v8f acc[2][2] = {};
  for (int k0 = 0; k0 < K; k0 += 32) {
    v16h a0 = load_a_frag<ADD2>(A, A2, K, row0, k0);
    v16h a1 = load_a_frag<ADD2>(A, A2, K, row0 + 16, k0);
    v16h b0 = load_b_frag(W, N, k0, col0);
    v16h b1 = load_b_frag(W, N, k0, col0 + 16);
    acc[0][0] = wmma32(a0, b0, acc[0][0]);
    acc[0][1] = wmma32(a0, b1, acc[0][1]);
    acc[1][0] = wmma32(a1, b0, acc[1][0]);
    acc[1][1] = wmma32(a1, b1, acc[1][1]);
  }
#pragma unroll
  for (int ti = 0; ti < 2; ++ti)
#pragma unroll
    for (int tj = 0; tj < 2; ++tj)
#pragma unroll
      for (int r = 0; r < 8; ++r) {
        const int row = row0 + ti * 16 + r + hi * 8;
        const int col = col0 + tj * 16 + n;
        float v = acc[ti][tj][r];
        if (EPI == 1) {  // sinusoidal positional encoding
          const int pos = row & (S_ - 1);
          const float dv =
              expf(-9.210340371976184f * (float)(col & ~1) / (float)D_);
          const float arg = (float)pos * dv;
          v += (col & 1) ? cosf(arg) : sinf(arg);
        } else if (EPI == 2) {
          v += bias[col];
        }
        C[(size_t)row * N + col] = v;
      }
}

// ---------------- Flash-style MLA attention with multiplicative decay ---------
// One wave handles 16 queries of one (b,h); streams keys in tiles of 32 with
// online softmax. scores = (q·k)/sqrt(D) * exp(-0.01*|i-j|).
__global__ __launch_bounds__(256) void mhla_attn(
    const float* __restrict__ q, const float* __restrict__ kr,
    const float* __restrict__ vr, float* __restrict__ out) {
  __shared__ _Float16 pbuf[8][16][34];  // per-wave 16x32 prob tile (+pad)
  const int Ln = threadIdx.x & 31;
  const int wv = threadIdx.x >> 5;
  const int n = Ln & 15, hi = (Ln >> 4) & 1;
  const int g = blockIdx.x * 8 + wv;      // global wave id: 0..2047
  const int qblk = g & (S_ / 16 - 1);     // 64 query blocks
  const int bh = g >> 6;
  const int hh = bh & (H_ - 1);
  const int bb = bh >> 4;

  const float* qp = q + ((size_t)bb * S_ + (size_t)qblk * 16) * D_ + hh * HD_;
  const float* kp = kr + (size_t)bb * S_ * D_ + hh * HD_;
  const float* vp = vr + (size_t)bb * S_ * D_ + hh * HD_;

  const v16h qa0 = load_a_frag<false>(qp, nullptr, D_, 0, 0);
  const v16h qa1 = load_a_frag<false>(qp, nullptr, D_, 0, 32);

  v8f o[4] = {};
  float mrow[8], lrow[8];
#pragma unroll
  for (int r = 0; r < 8; ++r) { mrow[r] = -1e30f; lrow[r] = 0.f; }

  for (int jb = 0; jb < S_; jb += 32) {
    v8f s0 = {}, s1 = {};
    s0 = wmma32(qa0, load_bT_frag(kp, D_, jb, 0), s0);
    s0 = wmma32(qa1, load_bT_frag(kp, D_, jb, 32), s0);
    s1 = wmma32(qa0, load_bT_frag(kp, D_, jb + 16, 0), s1);
    s1 = wmma32(qa1, load_bT_frag(kp, D_, jb + 16, 32), s1);

#pragma unroll
    for (int r = 0; r < 8; ++r) {
      const float qi = (float)(qblk * 16 + r + hi * 8);
      float t0 = s0[r] * 0.03125f * expf(-0.01f * fabsf(qi - (float)(jb + n)));
      float t1 =
          s1[r] * 0.03125f * expf(-0.01f * fabsf(qi - (float)(jb + 16 + n)));
      float tm = fmaxf(t0, t1);
#pragma unroll
      for (int x = 1; x < 16; x <<= 1) tm = fmaxf(tm, __shfl_xor(tm, x, 32));
      const float mnew = fmaxf(mrow[r], tm);
      const float corr = expf(mrow[r] - mnew);
      const float p0 = expf(t0 - mnew);
      const float p1 = expf(t1 - mnew);
      float ps = p0 + p1;
#pragma unroll
      for (int x = 1; x < 16; x <<= 1) ps += __shfl_xor(ps, x, 32);
      lrow[r] = lrow[r] * corr + ps;
      mrow[r] = mnew;
#pragma unroll
      for (int t = 0; t < 4; ++t) o[t][r] *= corr;
      pbuf[wv][r + hi * 8][n] = (_Float16)p0;
      pbuf[wv][r + hi * 8][16 + n] = (_Float16)p1;
    }
    __syncthreads();  // C-layout -> A-layout via per-wave LDS bounce
    v16h pa;
#pragma unroll
    for (int e = 0; e < 16; ++e) {
      const int Kk = (hi << 4) | e;
      pa[e] = pbuf[wv][Ln & 15][Kk];
    }
    __syncthreads();
#pragma unroll
    for (int t = 0; t < 4; ++t)
      o[t] = wmma32(pa, load_b_frag(vp, D_, jb, t * 16), o[t]);
  }

  float* op = out + ((size_t)bb * S_ + (size_t)qblk * 16) * D_ + hh * HD_;
#pragma unroll
  for (int t = 0; t < 4; ++t)
#pragma unroll
    for (int r = 0; r < 8; ++r)
      op[(size_t)(r + hi * 8) * D_ + t * 16 + n] = o[t][r] / lrow[r];
}

// ------------------------------- launcher -------------------------------------
extern "C" void kernel_launch(void* const* d_in, const int* in_sizes, int n_in,
                              void* d_out, int out_size, void* d_ws,
                              size_t ws_size, hipStream_t stream) {
  (void)in_sizes; (void)n_in; (void)out_size; (void)ws_size;
  const int*   src      = (const int*)d_in[0];
  // d_in[1] = tgt (unused by reference output)
  const float* a_n      = (const float*)d_in[2];
  const float* b_n      = (const float*)d_in[3];
  const float* proj_w   = (const float*)d_in[4];
  const float* proj_b   = (const float*)d_in[5];
  const float* rule_mat = (const float*)d_in[6];
  const float* Wq       = (const float*)d_in[7];
  const float* Wk       = (const float*)d_in[8];
  const float* Wv       = (const float*)d_in[9];
  const float* Wo       = (const float*)d_in[10];
  const float* Wdown    = (const float*)d_in[11];
  const float* Wup_k    = (const float*)d_in[12];
  const float* Wup_v    = (const float*)d_in[13];
  const float* Wout     = (const float*)d_in[14];
  const float* bout     = (const float*)d_in[15];
  float* outp = (float*)d_out;

  const size_t BS = (size_t)B_ * S_;
  float* ws   = (float*)d_ws;
  float* base = ws; ws += BS * D_;
  float* hbuf = ws; ws += BS * D_;
  float* qbuf = ws; ws += BS * D_;
  float* kbuf = ws; ws += BS * D_;
  float* vbuf = ws; ws += BS * D_;
  float* lat  = ws; ws += BS * DL_;
  float* krb  = ws; ws += BS * D_;
  float* vrb  = ws; ws += BS * D_;
  float* abuf = ws; ws += BS * D_;

  const dim3 blk(256);
  const int M = B_ * S_;
  auto gg = [](int Mm, int Nn) { return dim3(Nn / 128, Mm / 64); };

  embed_base_kernel<<<dim3(B_ * S_), blk, 0, stream>>>(src, a_n, b_n, proj_w,
                                                       proj_b, base);
  // h0 = base @ rule_mat + PE
  wmma_gemm<1, false><<<gg(M, D_), blk, 0, stream>>>(base, nullptr, rule_mat,
                                                     nullptr, hbuf, D_, D_);
  for (int l = 0; l < L_; ++l) {
    const size_t oDD = (size_t)l * D_ * D_;
    const size_t oDL = (size_t)l * D_ * DL_;
    wmma_gemm<0, false><<<gg(M, D_), blk, 0, stream>>>(
        hbuf, nullptr, Wq + oDD, nullptr, qbuf, D_, D_);
    wmma_gemm<0, false><<<gg(M, D_), blk, 0, stream>>>(
        hbuf, nullptr, Wk + oDD, nullptr, kbuf, D_, D_);
    wmma_gemm<0, false><<<gg(M, D_), blk, 0, stream>>>(
        hbuf, nullptr, Wv + oDD, nullptr, vbuf, D_, D_);
    // latent = (k + v) @ Wdown
    wmma_gemm<0, true><<<gg(M, DL_), blk, 0, stream>>>(
        kbuf, vbuf, Wdown + oDL, nullptr, lat, DL_, D_);
    wmma_gemm<0, false><<<gg(M, D_), blk, 0, stream>>>(
        lat, nullptr, Wup_k + oDL, nullptr, krb, D_, DL_);
    wmma_gemm<0, false><<<gg(M, D_), blk, 0, stream>>>(
        lat, nullptr, Wup_v + oDL, nullptr, vrb, D_, DL_);
    mhla_attn<<<dim3((B_ * H_ * (S_ / 16)) / 8), blk, 0, stream>>>(qbuf, krb,
                                                                   vrb, abuf);
    wmma_gemm<0, false><<<gg(M, D_), blk, 0, stream>>>(
        abuf, nullptr, Wo + oDD, nullptr, hbuf, D_, D_);
  }
  // logits = h @ Wout + bout
  wmma_gemm<2, false><<<gg(M, V_), blk, 0, stream>>>(hbuf, nullptr, Wout, bout,
                                                     outp, V_, D_);
}
